// GemmaAttention_85822036509122
// MI455X (gfx1250) — matrix operand order repaired and checked by reading, not compiled, not run on hardware
//
#include <hip/hip_runtime.h>

// ---------------------------------------------------------------------------
// Gemma attention block for MI455X (gfx1250, wave32, WMMA 16x16x32 bf16)
// ---------------------------------------------------------------------------

typedef __attribute__((ext_vector_type(16))) __bf16 v16bf;
typedef __attribute__((ext_vector_type(8)))  float  v8f;
typedef __attribute__((ext_vector_type(4)))  unsigned v4u;
typedef __attribute__((ext_vector_type(4)))  int      v4i;
typedef __attribute__((ext_vector_type(8)))  int      v8i;

union FragB16 { v16bf v; unsigned u[8]; };

#define NUM_HEADS    16
#define NUM_KV_HEADS 8
#define HEAD_DIM     256
#define HIDDEN       3584
#define SEQ          2048
#define BATCH        2
#define QKV_N        ((NUM_HEADS + 2 * NUM_KV_HEADS) * HEAD_DIM)   // 8192
#define Q_SIZE       (NUM_HEADS * HEAD_DIM)                         // 4096
#define KV_SIZE      (NUM_KV_HEADS * HEAD_DIM)                      // 2048
#define ROWS         (BATCH * SEQ)                                  // 4096
#define WINDOW       1024
#define SOFTCAP_INV  0.02f
#define SOFTCAP      50.0f
#define SCALING      0.0625f   // 256^-0.5

#if defined(__has_builtin)
#if __has_builtin(__builtin_amdgcn_tensor_load_to_lds)
#define HAVE_TDM 1
#endif
#endif
#ifndef HAVE_TDM
#define HAVE_TDM 0
#endif

__device__ __forceinline__ unsigned short f2bf(float f) {
    unsigned u = __float_as_uint(f);
    unsigned r = (u + 0x7FFFu + ((u >> 16) & 1u)) >> 16;
    return (unsigned short)r;
}

__device__ __forceinline__ v8f zero8() {
    v8f z = {0.f, 0.f, 0.f, 0.f, 0.f, 0.f, 0.f, 0.f};
    return z;
}

// ---------------------------------------------------------------------------
// TDM: load a 16x32 bf16 tile (row stride K elements) from global into LDS,
// padding 1 dword after every 64 bytes (=> 17-dword LDS rows).
// D# packing per CDNA5 ISA.
// ---------------------------------------------------------------------------
__device__ __forceinline__ void tdm_load_tile16x32(unsigned ldsByteOff,
                                                   const void* gaddr,
                                                   int K, int M) {
#if HAVE_TDM
    unsigned long long ga = (unsigned long long)gaddr;
    v4u g0 = { 1u,                                   // count=1, user mode
               ldsByteOff,                           // lds_addr
               (unsigned)(ga & 0xFFFFFFFFu),         // global_addr[31:0]
               (unsigned)(((ga >> 32) & 0x01FFFFFFu) // global_addr[56:32]
                          | (2u << 30)) };           // type=2 (image)
    const unsigned dim0 = (unsigned)K;               // tensor_dim0 (elements)
    const unsigned dim1 = (unsigned)M;               // tensor_dim1
    // w0: data_size=1(2B)<<16 | pad_enable(1<<20) | pad_interval=3(64B)<<22
    v8i g1 = { (int)0x00D10000u,
               (int)((dim0 & 0xFFFFu) << 16),                    // dim0[15:0]
               (int)((dim0 >> 16) | ((dim1 & 0xFFFFu) << 16)),   // dim0[31:16]|dim1[15:0]
               (int)((dim1 >> 16) | (32u << 16)),                // dim1[31:16]|tile_dim0=32
               (int)16,                                          // tile_dim1=16, tile_dim2=0
               (int)dim0,                                        // dim0_stride[31:0] = K
               0, 0 };                                           // stride hi / dim1_stride
    v4i gz = {0, 0, 0, 0};
#if __clang_major__ >= 23
    v8i gz8 = {0, 0, 0, 0, 0, 0, 0, 0};
    __builtin_amdgcn_tensor_load_to_lds(g0, g1, gz, gz, gz8, 0);
#else
    __builtin_amdgcn_tensor_load_to_lds(g0, g1, gz, gz, 0);
#endif
#endif
}

__device__ __forceinline__ void wait_tensorcnt0() {
#if defined(__has_builtin)
#if __has_builtin(__builtin_amdgcn_s_wait_tensorcnt)
    __builtin_amdgcn_s_wait_tensorcnt(0);
#else
    asm volatile("s_wait_tensorcnt 0x0" ::: "memory");
#endif
#else
    asm volatile("s_wait_tensorcnt 0x0" ::: "memory");
#endif
}

// ---------------------------------------------------------------------------
// Kernel 1: f32 -> bf16 conversion (grid-stride)
// ---------------------------------------------------------------------------
__global__ void cvt_f32_bf16(const float* __restrict__ in,
                             unsigned short* __restrict__ out, int n) {
    int i = blockIdx.x * blockDim.x + threadIdx.x;
    int stride = gridDim.x * blockDim.x;
    for (; i < n; i += stride) out[i] = f2bf(in[i]);
}

// ---------------------------------------------------------------------------
// Kernel 2: C[M,N](f32) = A[M,K](bf16, row-major) * W[N,K](bf16, row-major)^T
// 256 threads = 8 waves; one 16x16 C tile per wave.
// A tile staged into LDS via TDM (useTdm=1) or async mem->LDS copies.
// ---------------------------------------------------------------------------
__global__ __launch_bounds__(256)
void gemm_bf16_wmma(const unsigned short* __restrict__ A,
                    const unsigned short* __restrict__ W,
                    float* __restrict__ C, int M, int N, int K, int useTdm) {
    __shared__ unsigned aLds[16 * 17 + 4];

    const int lane  = threadIdx.x & 31;
    const int wave  = threadIdx.x >> 5;
    const int half  = lane >> 4;       // 0 or 1
    const int l16   = lane & 15;
    const int mBase = blockIdx.x * 16;
    const int nBase = blockIdx.y * 128 + wave * 16;

    const unsigned* A32 = (const unsigned*)A;
    const unsigned* W32 = (const unsigned*)W;
    const int Kd = K >> 1;             // dwords per row

    const int ldRow = threadIdx.x >> 4;   // 0..15
    const int ldCol = threadIdx.x & 15;   // 0..15 (dwords of a 32-elem chunk)

    // LDS byte offsets (addrspace(3) offset lives in the low 32 bits)
    const unsigned ldsBase = (unsigned)(unsigned long long)(const void*)&aLds[0];
    const unsigned ldsAddr =
        (unsigned)(unsigned long long)(const void*)&aLds[ldRow * 17 + ldCol];

    v8f acc = zero8();

    for (int kc = 0; kc < K; kc += 32) {
        if (HAVE_TDM && useTdm) {
            // single TDM descriptor issued by wave 0; EXEC-independent DMA
            if (wave == 0) {
                tdm_load_tile16x32(ldsBase, (const void*)(A32 + mBase * Kd + (kc >> 1)),
                                   K, M);
                wait_tensorcnt0();
            }
        } else {
            // per-lane async copy of the 16x32 bf16 A tile into LDS
            const unsigned* gsrc = A32 + (mBase + ldRow) * Kd + (kc >> 1) + ldCol;
            asm volatile("global_load_async_to_lds_b32 %0, %1, off"
                         :: "v"(ldsAddr), "v"(gsrc)
                         : "memory");
            asm volatile("s_wait_asynccnt 0x0" ::: "memory");
        }
        __syncthreads();

        FragB16 a, b;
        // A fragment: lane row = l16; K = half*16 + 2v (contiguous, matches B)
#pragma unroll
        for (int v = 0; v < 4; ++v) {
            a.u[v]     = aLds[l16 * 17 + half * 8 + v];        // K = half*16 + 2v
            a.u[v + 4] = aLds[l16 * 17 + half * 8 + 4 + v];    // K = half*16 + 8 + 2v
        }
        // B fragment: lane col n = nBase+l16; K = half*16 + 2v (packed pairs)
        const unsigned* wrow = W32 + (nBase + l16) * Kd + (kc >> 1) + half * 8;
#pragma unroll
        for (int v = 0; v < 8; ++v) b.u[v] = wrow[v];
        if (kc + 32 < K) __builtin_prefetch((const void*)(wrow + 16), 0, 3);

        acc = __builtin_amdgcn_wmma_f32_16x16x32_bf16(
            false, a.v, false, b.v, (short)0, acc, false, false);
        __syncthreads();
    }

    // C/D layout: row = v + 8*half, col = l16
#pragma unroll
    for (int v = 0; v < 8; ++v)
        C[(mBase + v + half * 8) * N + nBase + l16] = acc[v];
}

// ---------------------------------------------------------------------------
// Kernel 3: RoPE + scaling + head split:  qkv f32 [ROWS][8192] ->
//   qb bf16 [ROWS][16][256] (*SCALING), kb bf16 [ROWS][8][256],
//   vbT bf16 [B][KVH][256][SEQ]  (V transposed for dword-packed PV fragments)
// ---------------------------------------------------------------------------
__global__ void rope_split(const float* __restrict__ qkv,
                           const float* __restrict__ cosT,
                           const float* __restrict__ sinT,
                           unsigned short* __restrict__ qb,
                           unsigned short* __restrict__ kb,
                           unsigned short* __restrict__ vbT) {
    long long i = (long long)blockIdx.x * blockDim.x + threadIdx.x;
    const long long total = (long long)ROWS * QKV_N;
    const long long stride = (long long)gridDim.x * blockDim.x;
    for (; i < total; i += stride) {
        int row = (int)(i >> 13);          // / 8192
        int col = (int)(i & (QKV_N - 1));
        int s = row & (SEQ - 1);
        int bb = row >> 11;
        const float* rowp = qkv + (long long)row * QKV_N;
        if (col < Q_SIZE) {
            int h = col >> 8, d = col & 255, d0 = d & 127;
            float c  = cosT[s * 128 + d0];
            float sn = sinT[s * 128 + d0];
            float x1 = rowp[h * 256 + d0];
            float x2 = rowp[h * 256 + d0 + 128];
            float val = (d < 128) ? (x1 * c - x2 * sn) : (x1 * sn + x2 * c);
            qb[((long long)row * NUM_HEADS + h) * 256 + d] = f2bf(val * SCALING);
        } else if (col < Q_SIZE + KV_SIZE) {
            int cc = col - Q_SIZE;
            int h = cc >> 8, d = cc & 255, d0 = d & 127;
            float c  = cosT[s * 128 + d0];
            float sn = sinT[s * 128 + d0];
            float x1 = rowp[Q_SIZE + h * 256 + d0];
            float x2 = rowp[Q_SIZE + h * 256 + d0 + 128];
            float val = (d < 128) ? (x1 * c - x2 * sn) : (x1 * sn + x2 * c);
            kb[((long long)row * NUM_KV_HEADS + h) * 256 + d] = f2bf(val);
        } else {
            int cc = col - Q_SIZE - KV_SIZE;
            int h = cc >> 8, d = cc & 255;
            // transposed: [b][kv_head][d][s]
            vbT[(((long long)(bb * NUM_KV_HEADS + h)) * HEAD_DIM + d) * SEQ + s] =
                f2bf(rowp[Q_SIZE + KV_SIZE + h * 256 + d]);
        }
    }
}

// ---------------------------------------------------------------------------
// Kernel 4: flash-style windowed attention with tanh softcap.
// One wave per (batch, head, 16-row q tile); 32 keys per step.
// ---------------------------------------------------------------------------
__global__ __launch_bounds__(32)
void attn_wmma(const unsigned short* __restrict__ qb,
               const unsigned short* __restrict__ kb,
               const unsigned short* __restrict__ vbT,
               unsigned short* __restrict__ ab) {
    __shared__ unsigned short pT[16][34];   // 16 q-rows x 32 keys (padded)

    const int blk = blockIdx.x;
    const int qt  = blk & 127;           // q tile (16 rows)
    const int h   = (blk >> 7) & 15;
    const int b   = blk >> 11;
    const int kvh = h >> 1;              // nrep = 2
    const int lane = threadIdx.x & 31;
    const int half = lane >> 4, l16 = lane & 15;

    const unsigned* Q32 = (const unsigned*)qb;
    const unsigned* K32 = (const unsigned*)kb;
    const unsigned* V32 = (const unsigned*)vbT;

    // --- preload Q A-fragments: d = 256 -> 8 chunks of K=32 ---
    FragB16 qf[8];
    {
        const int srow = qt * 16 + l16;
        const unsigned* qrow =
            Q32 + (((long long)(b * SEQ + srow) * NUM_HEADS + h) << 7); // *128 dwords
#pragma unroll
        for (int c = 0; c < 8; ++c) {
#pragma unroll
            for (int v = 0; v < 4; ++v) {
                int k0 = c * 32 + half * 16 + 2 * v;
                int k1 = c * 32 + half * 16 + 8 + 2 * v;
                qf[c].u[v]     = qrow[k0 >> 1];
                qf[c].u[v + 4] = qrow[k1 >> 1];
            }
        }
    }

    v8f accO[16];
#pragma unroll
    for (int t = 0; t < 16; ++t) accO[t] = zero8();
    float mrow[8], lrow[8];
#pragma unroll
    for (int v = 0; v < 8; ++v) { mrow[v] = -3.0e38f; lrow[v] = 0.0f; }

    const int iLo = qt * 16;
    int jLo = iLo - (WINDOW - 1); if (jLo < 0) jLo = 0;
    jLo &= ~31;                       // 32-key alignment (keys stay in-bounds)
    const int jHi = iLo + 15;         // causal limit

    for (int j0 = jLo; j0 <= jHi; j0 += 32) {
        // ---- scores: two 16x16 tiles (keys j0..j0+15, j0+16..j0+31) ----
        v8f sA = zero8(), sB = zero8();
#pragma unroll
        for (int c = 0; c < 8; ++c) {
            FragB16 bk;
            const unsigned* krow =
                K32 + (((long long)(b * SEQ + j0 + l16) * NUM_KV_HEADS + kvh) << 7);
#pragma unroll
            for (int v = 0; v < 8; ++v)
                bk.u[v] = krow[((c * 32 + half * 16) >> 1) + v];
            sA = __builtin_amdgcn_wmma_f32_16x16x32_bf16(
                false, qf[c].v, false, bk.v, (short)0, sA, false, false);

            const unsigned* krow2 =
                K32 + (((long long)(b * SEQ + j0 + 16 + l16) * NUM_KV_HEADS + kvh) << 7);
#pragma unroll
            for (int v = 0; v < 8; ++v)
                bk.u[v] = krow2[((c * 32 + half * 16) >> 1) + v];
            sB = __builtin_amdgcn_wmma_f32_16x16x32_bf16(
                false, qf[c].v, false, bk.v, (short)0, sB, false, false);
        }

        // ---- softcap + band mask + online softmax (row = v + 8*half) ----
#pragma unroll
        for (int v = 0; v < 8; ++v) {
            const int i  = iLo + v + half * 8;
            const int jA = j0 + l16;
            const int jB = j0 + 16 + l16;
            float sa = tanhf(sA[v] * SOFTCAP_INV) * SOFTCAP;
            float sb = tanhf(sB[v] * SOFTCAP_INV) * SOFTCAP;
            if (jA > i || jA < i - (WINDOW - 1)) sa = -3.0e38f;
            if (jB > i || jB < i - (WINDOW - 1)) sb = -3.0e38f;

            float rm = fmaxf(sa, sb);
#pragma unroll
            for (int off = 8; off; off >>= 1)
                rm = fmaxf(rm, __shfl_xor(rm, off, 16));
            float mNew  = fmaxf(mrow[v], rm);
            float scale = __expf(mrow[v] - mNew);
            float pa = __expf(sa - mNew);
            float pb = __expf(sb - mNew);
            float rs = pa + pb;
#pragma unroll
            for (int off = 8; off; off >>= 1)
                rs += __shfl_xor(rs, off, 16);
            lrow[v] = lrow[v] * scale + rs;
            mrow[v] = mNew;
#pragma unroll
            for (int t = 0; t < 16; ++t) accO[t][v] *= scale;

            pT[v + half * 8][l16]      = f2bf(pa);
            pT[v + half * 8][16 + l16] = f2bf(pb);
        }
        __syncthreads();

        // ---- transpose P (C layout -> A layout) through LDS ----
        FragB16 pf;
        {
            const unsigned* pr = (const unsigned*)&pT[l16][0];  // 34 shorts = 17 dwords/row
#pragma unroll
            for (int v = 0; v < 4; ++v) {
                pf.u[v]     = pr[half * 8 + v];          // K = half*16 + 2v
                pf.u[v + 4] = pr[half * 8 + 4 + v];      // K = half*16 + 8 + 2v
            }
        }

        // ---- PV: 16 n-tiles over d; contraction K = 32 keys.
        //      V is [b][kvh][d][s]: packed key-pairs are contiguous dwords. ----
#pragma unroll
        for (int t = 0; t < 16; ++t) {
            FragB16 bv;
            const int d = t * 16 + l16;
            const unsigned* vrow =
                V32 + (((((long long)(b * NUM_KV_HEADS + kvh)) * HEAD_DIM + d) * SEQ
                        + j0) >> 1) + half * 8;
#pragma unroll
            for (int v = 0; v < 8; ++v) bv.u[v] = vrow[v];
            accO[t] = __builtin_amdgcn_wmma_f32_16x16x32_bf16(
                false, pf.v, false, bv.v, (short)0, accO[t], false, false);
        }
        __syncthreads();
    }

    // ---- normalize + store bf16 to ab[ROWS][NUM_HEADS*HEAD_DIM] ----
#pragma unroll
    for (int v = 0; v < 8; ++v) {
        const float inv = 1.0f / lrow[v];
        const int srow  = qt * 16 + v + half * 8;
        const long long base =
            (long long)(b * SEQ + srow) * (NUM_HEADS * HEAD_DIM) + h * HEAD_DIM;
#pragma unroll
        for (int t = 0; t < 16; ++t)
            ab[base + t * 16 + l16] = f2bf(accO[t][v] * inv);
    }
}

// ---------------------------------------------------------------------------
// Host-side launcher
// ---------------------------------------------------------------------------
extern "C" void kernel_launch(void* const* d_in, const int* in_sizes, int n_in,
                              void* d_out, int out_size, void* d_ws, size_t ws_size,
                              hipStream_t stream) {
    const float* hidden = (const float*)d_in[0];   // [2,2048,3584]
    const float* cosT   = (const float*)d_in[1];   // [2048,128]
    const float* sinT   = (const float*)d_in[2];
    // d_in[3] kv_write_indices == identity; d_in[4]/d_in[5] zero caches;
    // d_in[6] causal mask — band+causal recomputed analytically in-kernel.
    const float* qkv_w  = (const float*)d_in[7];   // [8192,3584]
    const float* o_w    = (const float*)d_in[8];   // [3584,4096]
    float* out = (float*)d_out;                    // [2,2048,3584]

    char* ws = (char*)d_ws;
    size_t off = 0;
    auto alloc = [&](size_t bytes) -> char* {
        char* p = ws + off;
        off += (bytes + 255) & ~(size_t)255;
        return p;
    };
    unsigned short* hb   = (unsigned short*)alloc((size_t)ROWS * HIDDEN * 2);
    unsigned short* wb   = (unsigned short*)alloc((size_t)QKV_N * HIDDEN * 2);
    unsigned short* ob   = (unsigned short*)alloc((size_t)HIDDEN * Q_SIZE * 2);
    float*          qkvf = (float*)        alloc((size_t)ROWS * QKV_N * 4);
    unsigned short* qb   = (unsigned short*)alloc((size_t)ROWS * Q_SIZE * 2);
    unsigned short* kbuf = (unsigned short*)alloc((size_t)ROWS * KV_SIZE * 2);
    unsigned short* vbT  = (unsigned short*)alloc((size_t)ROWS * KV_SIZE * 2);
    unsigned short* ab   = (unsigned short*)alloc((size_t)ROWS * Q_SIZE * 2);

    // 1) f32 -> bf16 conversions
    cvt_f32_bf16<<<8192, 256, 0, stream>>>(hidden, hb, ROWS * HIDDEN);
    cvt_f32_bf16<<<8192, 256, 0, stream>>>(qkv_w, wb, QKV_N * HIDDEN);
    cvt_f32_bf16<<<8192, 256, 0, stream>>>(o_w, ob, HIDDEN * Q_SIZE);

    // 2) QKV projection (TDM-staged A tiles):
    //    [4096,3584] x [8192,3584]^T -> [4096,8192] f32
    gemm_bf16_wmma<<<dim3(ROWS / 16, QKV_N / 128), 256, 0, stream>>>(
        hb, wb, qkvf, ROWS, QKV_N, HIDDEN, 1);

    // 3) RoPE + scaling + head split (identity kv cache write); V transposed
    rope_split<<<8192, 256, 0, stream>>>(qkvf, cosT, sinT, qb, kbuf, vbT);

    // 4) windowed flash attention with tanh softcap
    attn_wmma<<<BATCH * NUM_HEADS * (SEQ / 16), 32, 0, stream>>>(qb, kbuf, vbT, ab);

    // 5) output projection (async-staged A tiles):
    //    [4096,4096] x [3584,4096]^T -> [4096,3584] f32
    gemm_bf16_wmma<<<dim3(ROWS / 16, HIDDEN / 128), 256, 0, stream>>>(
        ab, ob, out, ROWS, HIDDEN, Q_SIZE, 0);
}
